// SelfAttention_7189775253842
// MI455X (gfx1250) — matrix operand order repaired and checked by reading, not compiled, not run on hardware
//
#include <hip/hip_runtime.h>

// ---- problem constants (from reference) ----
constexpr int kB  = 32;
constexpr int kT  = 2048;
constexpr int kD  = 384;
constexpr int kHS = 64;

typedef __bf16 bf16;
typedef __attribute__((ext_vector_type(16))) bf16  v16bf;
typedef __attribute__((ext_vector_type(8)))  bf16  bf16x8;
typedef __attribute__((ext_vector_type(8)))  float v8f;
typedef __attribute__((ext_vector_type(4)))  float f32x4;

__device__ __forceinline__ v8f wmma_bf16(v16bf a, v16bf b, v8f c) {
  // D(f32 16x16) = A(bf16 16x32) * B(bf16 32x16) + C
  return __builtin_amdgcn_wmma_f32_16x16x32_bf16(
      /*neg_a=*/false, a, /*neg_b=*/false, b,
      /*c_mod=*/(short)0, c, /*reuse_a=*/false, /*reuse_b=*/false);
}

// =====================================================================
// Phase 0: repack W -> Wt[3][64][384] bf16 (transposed, Q pre-scaled by
// HS^-0.5) so phase-1 B-fragments are two contiguous b128 loads per lane.
// =====================================================================
__global__ __launch_bounds__(256, 1) void wt_convert_kernel(
    const float* __restrict__ Wq, const float* __restrict__ Wk,
    const float* __restrict__ Wv, bf16* __restrict__ wt)
{
  const int gid = blockIdx.x * 256 + threadIdx.x;     // 0 .. 3*64*384-1
  if (gid >= 3 * kHS * kD) return;
  const int mode = gid / (kHS * kD);
  const int rem  = gid - mode * (kHS * kD);
  const int n    = rem / kD;                          // output column 0..63
  const int kx   = rem - n * kD;                      // reduction index 0..383
  const float* W = (mode == 0) ? Wq : (mode == 1) ? Wk : Wv;
  const float s  = (mode == 0) ? 0.125f : 1.0f;       // 1/sqrt(64) folded into Wq
  wt[gid] = (bf16)(W[(size_t)kx * kHS + n] * s);
}

// =====================================================================
// Phase 1 (fused QKV): one wave = one 16-row x 16-col tile, computing the
// Q, K and V tiles from a single A-fragment (x read once, 3 WMMAs/chunk).
//   q,k -> [B*T, HS] bf16 row-major;  v -> [B, HS, T] bf16 (transposed)
// =====================================================================
__global__ __launch_bounds__(128, 1) void qkv_proj_kernel(
    const float* __restrict__ x,
    const bf16* __restrict__ wt,      // [3][64][384]
    bf16* __restrict__ qout,
    bf16* __restrict__ kout,
    bf16* __restrict__ vtout)
{
  const int lane = threadIdx.x & 31;
  const int wave = threadIdx.x >> 5;
  const int half = lane >> 4;
  const int l16  = lane & 15;
  const int kb   = half * 8;

  const int m0 = blockIdx.x * 16;     // token tile
  const int n0 = wave * 16;           // HS column tile

  const float* __restrict__ xrow = x + (size_t)(m0 + l16) * kD;
  const bf16*  __restrict__ wcol = wt + (size_t)(n0 + l16) * kD;

  v8f acc0 = {}, acc1 = {}, acc2 = {};
#pragma unroll
  for (int kk = 0; kk < kD; kk += 32) {
    // A fragment: x tile 16x32 fp32 -> bf16 (two contiguous 32B runs/lane).
    const f32x4* pa0 = (const f32x4*)(xrow + kk + kb);
    const f32x4* pa1 = (const f32x4*)(xrow + kk + kb + 16);
    f32x4 a0 = pa0[0], a1 = pa0[1];
    f32x4 a2 = pa1[0], a3 = pa1[1];
    v16bf afrag;
#pragma unroll
    for (int j = 0; j < 4; ++j) {
      afrag[j]      = (bf16)a0[j];
      afrag[4 + j]  = (bf16)a1[j];
      afrag[8 + j]  = (bf16)a2[j];
      afrag[12 + j] = (bf16)a3[j];
    }
    // B fragments from Wt: two 8-bf16 runs (k = kb.., kb+16..) per lane per matrix.
    v16bf bq, bk2, bv;
#pragma unroll
    for (int m = 0; m < 3; ++m) {
      const bf16* wp = wcol + (size_t)m * kHS * kD + kk + half * 8;
      bf16x8 r0 = *(const bf16x8*)(wp);
      bf16x8 r1 = *(const bf16x8*)(wp + 16);
      v16bf bf;
#pragma unroll
      for (int j = 0; j < 8; ++j) { bf[j] = r0[j]; bf[8 + j] = r1[j]; }
      if (m == 0) bq = bf; else if (m == 1) bk2 = bf; else bv = bf;
    }
    acc0 = wmma_bf16(afrag, bq,  acc0);
    acc1 = wmma_bf16(afrag, bk2, acc1);
    acc2 = wmma_bf16(afrag, bv,  acc2);
  }

  // C layout: VGPR g -> row m0+g+half*8, col n0+l16.
  const int col = n0 + l16;
#pragma unroll
  for (int g = 0; g < 8; ++g) {
    const int row = m0 + g + half * 8;
    qout[(size_t)row * kHS + col] = (bf16)acc0[g];
    kout[(size_t)row * kHS + col] = (bf16)acc1[g];
  }
  // V transposed: per lane the 8 rows are CONSECUTIVE t -> one b128 store.
  const int bb = m0 >> 11;                    // batch
  const int t0 = (m0 & (kT - 1)) + half * 8;  // token within batch
  bf16x8 vv;
#pragma unroll
  for (int g = 0; g < 8; ++g) vv[g] = (bf16)acc2[g];
  *(bf16x8*)(vtout + ((size_t)bb * kHS + col) * kT + t0) = vv;
}

// =====================================================================
// Phase 2: causal flash attention, block-cooperative double-buffered
// K/V staging via GLOBAL_LOAD_ASYNC_TO_LDS_B128 (ASYNCcnt) with workgroup
// split-barriers. Per wave: 16-query tile; per 32-key block: 4 WMMA for
// S = QK^T, online softmax (shfl reductions), P relayout via LDS, 4 WMMA
// for O += P*V.
// =====================================================================
__global__ __launch_bounds__(128, 1) void flash_attn_kernel(
    const bf16* __restrict__ q,    // [B*T, HS], pre-scaled
    const bf16* __restrict__ k,    // [B*T, HS]
    const bf16* __restrict__ vt,   // [B, HS, T]
    float* __restrict__ out)       // [B, T, HS] fp32
{
  __shared__ __align__(16) bf16 ksh[2][32][64];     // [buf][key][hs]  4KB/buf
  __shared__ __align__(16) bf16 vsh[2][64][32];     // [buf][hs][key]  4KB/buf
  __shared__ __align__(16) bf16 pstage[4][16][32];  // [wave][row][key]

  const int tid  = threadIdx.x;
  const int lane = tid & 31;
  const int wave = tid >> 5;
  const int half = lane >> 4;
  const int l16  = lane & 15;
  const int kb   = half * 8;

  const int b  = blockIdx.x;
  const int q0 = blockIdx.y * 64 + wave * 16;

  // --- Q A-fragments for HS chunks [0,32) and [32,64) ---
  const bf16* __restrict__ qrow = q + (size_t)(b * kT + q0 + l16) * kHS;
  v16bf aq0, aq1;
  {
    bf16x8 r0 = *(const bf16x8*)(qrow + kb);
    bf16x8 r1 = *(const bf16x8*)(qrow + kb + 16);
    bf16x8 r2 = *(const bf16x8*)(qrow + 32 + kb);
    bf16x8 r3 = *(const bf16x8*)(qrow + 32 + kb + 16);
#pragma unroll
    for (int j = 0; j < 8; ++j) {
      aq0[j] = r0[j]; aq0[8 + j] = r1[j];
      aq1[j] = r2[j]; aq1[8 + j] = r3[j];
    }
  }

  v8f o[4] = {};
  float mrow[8], lrow[8];
#pragma unroll
  for (int g = 0; g < 8; ++g) { mrow[g] = -__builtin_inff(); lrow[g] = 0.0f; }

  const bf16* __restrict__ kbase = k  + (size_t)b * kT * kHS;
  const bf16* __restrict__ vbase = vt + (size_t)b * kHS * kT;

  const unsigned ldsk = (unsigned)(uintptr_t)(&ksh[0][0][0]);
  const unsigned ldsv = (unsigned)(uintptr_t)(&vsh[0][0][0]);

  // Cooperative async stage of one 32-key K tile (contiguous 4KB) and the
  // matching V tile (64 x 64B strided rows) into LDS buffer `buf`.
  auto stage = [&](int buf, int kb0) {
    const char* gk = (const char*)(kbase + (size_t)kb0 * kHS);
    const char* gv = (const char*)vbase + (size_t)kb0 * 2;
    const unsigned lk = ldsk + buf * 4096;
    const unsigned lv = ldsv + buf * 4096;
#pragma unroll
    for (int t = 0; t < 2; ++t) {
      const int idx = tid * 16 + t * 2048;          // byte index within 4KB tile
      unsigned long long gak = (unsigned long long)(uintptr_t)(gk + idx);
      asm volatile("global_load_async_to_lds_b128 %0, %1, off"
                   :: "v"(lk + idx), "v"(gak) : "memory");
      const int row = idx >> 6;                     // hs row of V tile
      const int cb  = idx & 63;                     // byte within 64B row
      unsigned long long gav =
          (unsigned long long)(uintptr_t)(gv + (size_t)row * (kT * 2) + cb);
      asm volatile("global_load_async_to_lds_b128 %0, %1, off"
                   :: "v"(lv + idx), "v"(gav) : "memory");
    }
  };

  // Uniform trip count across the block; early waves mask surplus keys.
  const int nblk = blockIdx.y * 2 + 2;

  stage(0, 0);
  asm volatile("s_wait_asynccnt 0" ::: "memory");
  __syncthreads();

  for (int i = 0; i < nblk; ++i) {
    const int cur = i & 1;
    const int kb0 = i * 32;
    if (i + 1 < nblk) stage(cur ^ 1, kb0 + 32);     // prefetch next block

    // ---- S = Q K^T for two 16-key tiles (fragments from LDS) ----
    v8f s0 = {}, s1 = {};
#pragma unroll
    for (int t = 0; t < 2; ++t) {
      const bf16* kr = &ksh[cur][t * 16 + l16][half * 8];
      bf16x8 r0 = *(const bf16x8*)(kr);
      bf16x8 r1 = *(const bf16x8*)(kr + 16);
      bf16x8 r2 = *(const bf16x8*)(kr + 32);
      bf16x8 r3 = *(const bf16x8*)(kr + 48);
      v16bf bk0, bk1;
#pragma unroll
      for (int j = 0; j < 8; ++j) {
        bk0[j] = r0[j]; bk0[8 + j] = r1[j];
        bk1[j] = r2[j]; bk1[8 + j] = r3[j];
      }
      if (t == 0) {
        s0 = wmma_bf16(aq0, bk0, s0);
        s0 = wmma_bf16(aq1, bk1, s0);
      } else {
        s1 = wmma_bf16(aq0, bk0, s1);
        s1 = wmma_bf16(aq1, bk1, s1);
      }
    }

    // ---- online softmax ----
    const int key0 = kb0 + l16;
    const int key1 = kb0 + 16 + l16;
#pragma unroll
    for (int g = 0; g < 8; ++g) {
      const int row = q0 + g + half * 8;
      float a = s0[g], c = s1[g];
      if (key0 > row) a = -__builtin_inff();
      if (key1 > row) c = -__builtin_inff();
      float bm = fmaxf(a, c);
#pragma unroll
      for (int off = 1; off < 16; off <<= 1)
        bm = fmaxf(bm, __shfl_xor(bm, off, 16));
      const float mn    = fmaxf(mrow[g], bm);
      const float alpha = __expf(mrow[g] - mn);
      const float p0    = __expf(a - mn);
      const float p1    = __expf(c - mn);
      float rs = p0 + p1;
#pragma unroll
      for (int off = 1; off < 16; off <<= 1)
        rs += __shfl_xor(rs, off, 16);
      lrow[g] = lrow[g] * alpha + rs;
      mrow[g] = mn;
#pragma unroll
      for (int n = 0; n < 4; ++n) o[n][g] *= alpha;

      const int prow = g + half * 8;
      pstage[wave][prow][l16]      = (bf16)p0;
      pstage[wave][prow][16 + l16] = (bf16)p1;
    }

    // Same-wave LDS ordering before the cross-lane relayout read.
    asm volatile("s_wait_dscnt 0" ::: "memory");

    // ---- reload P as a 16x32 A fragment ----
    v16bf ap;
    {
      const bf16* pr = &pstage[wave][l16][0];
      bf16x8 r0 = *(const bf16x8*)(pr + kb);
      bf16x8 r1 = *(const bf16x8*)(pr + kb + 16);
#pragma unroll
      for (int j = 0; j < 8; ++j) { ap[j] = r0[j]; ap[8 + j] = r1[j]; }
    }

    // ---- O += P * V over 4 HS tiles (V fragments from LDS) ----
#pragma unroll
    for (int n = 0; n < 4; ++n) {
      const bf16* vp = &vsh[cur][n * 16 + l16][half * 8];
      bf16x8 r0 = *(const bf16x8*)(vp);
      bf16x8 r1 = *(const bf16x8*)(vp + 16);
      v16bf bv;
#pragma unroll
      for (int j = 0; j < 8; ++j) { bv[j] = r0[j]; bv[8 + j] = r1[j]; }
      o[n] = wmma_bf16(ap, bv, o[n]);
    }

    // Prefetch landed? (own asyncs) + all waves done with `cur` before it is
    // overwritten next iteration.
    asm volatile("s_wait_asynccnt 0" ::: "memory");
    __syncthreads();
  }

  // ---- epilogue: O / l, fp32 out ----
  float* __restrict__ obase = out + (size_t)b * kT * kHS;
#pragma unroll
  for (int g = 0; g < 8; ++g) {
    const int row = q0 + g + half * 8;
    const float inv = 1.0f / lrow[g];
#pragma unroll
    for (int n = 0; n < 4; ++n)
      obase[(size_t)row * kHS + n * 16 + l16] = o[n][g] * inv;
  }
}

// =====================================================================
extern "C" void kernel_launch(void* const* d_in, const int* in_sizes, int n_in,
                              void* d_out, int out_size, void* d_ws, size_t ws_size,
                              hipStream_t stream) {
  const float* x  = (const float*)d_in[0];
  const float* Wq = (const float*)d_in[1];
  const float* Wk = (const float*)d_in[2];
  const float* Wv = (const float*)d_in[3];

  // Workspace: q (8 MB) | k (8 MB) | v-transposed (8 MB) | Wt (144 KB), bf16.
  const size_t n_qk = (size_t)kB * kT * kHS;
  bf16* qws  = (bf16*)d_ws;
  bf16* kws  = qws + n_qk;
  bf16* vtws = kws + n_qk;
  bf16* wtws = vtws + n_qk;

  wt_convert_kernel<<<dim3((3 * kHS * kD) / 256), 256, 0, stream>>>(
      Wq, Wk, Wv, wtws);
  qkv_proj_kernel<<<dim3(kB * kT / 16), 128, 0, stream>>>(
      x, wtws, qws, kws, vtws);
  flash_attn_kernel<<<dim3(kB, kT / 64), 128, 0, stream>>>(
      qws, kws, vtws, (float*)d_out);
}
